// BertConnectionLayer_15676630630432
// MI455X (gfx1250) — compile-verified, matrix operand order checked by reading, not run on hardware
//
#include <hip/hip_runtime.h>
#include <hip/hip_bf16.h>
#include <math.h>

// ---- problem constants (from reference) ----
#define B_    4
#define L_    512          // T*S
#define NTOK  2048         // B*L
#define DH_   768
#define BI_   1024
#define H_    16
#define HD_   64
#define HKV_  32
#define KV_   512
#define IM_   3072
#define EPS_  1e-12f
#define SCALE_ 0.125f      // 1/sqrt(64)

#define LDSP  40           // padded LDS row stride in halves (80B: 16B-aligned, bank-conflict-free)

typedef __attribute__((ext_vector_type(16))) _Float16 v16h;
typedef __attribute__((ext_vector_type(8)))  _Float16 v8h;
typedef __attribute__((ext_vector_type(8)))  float    v8f;

// Load one 16x32 f16 fragment (A-style / B-column-style) from a row-major
// [rows, ld] f16 matrix. Per ISA 7.12.2 layout: lane L holds row (base+L%16);
// element e maps to k = kbase + (e/8)*16 + (L/16)*8 + (e%8)  -> two contiguous
// 8-half (16B) chunks per lane. Works for global or LDS pointers.
__device__ __forceinline__ v16h ld_frag(const _Float16* base, int ld,
                                        int row, int kb, int lane) {
  const _Float16* p = base + (size_t)(row + (lane & 15)) * (size_t)ld
                           + (size_t)(kb + ((lane >> 4) << 3));
  v8h lo = *(const v8h*)p;
  v8h hi = *(const v8h*)(p + 16);
  return __builtin_shufflevector(lo, hi, 0,1,2,3,4,5,6,7,8,9,10,11,12,13,14,15);
}

__device__ __forceinline__ v8f wmma16(v16h a, v16h b, v8f c) {
  return __builtin_amdgcn_wmma_f32_16x16x32_f16(false, a, false, b, (short)0, c, false, false);
}

// CDNA5 async copy: 16B per lane, global -> LDS, tracked by ASYNCcnt.
__device__ __forceinline__ void async_cp16(unsigned lds_off, const void* gaddr) {
  asm volatile("global_load_async_to_lds_b128 %0, %1, off"
               :: "v"(lds_off), "v"((unsigned long long)(uintptr_t)gaddr)
               : "memory");
}
__device__ __forceinline__ void wait_async0() {
  asm volatile("s_wait_asynccnt 0x0" ::: "memory");
}

// ---------------- conversion kernels ----------------
__global__ void k_f32_to_f16(const float* __restrict__ src, _Float16* __restrict__ dst, int n) {
  int i = blockIdx.x * 256 + threadIdx.x;
  if (i < n) dst[i] = (_Float16)src[i];
}

// W[K,N] fp32  ->  WT[N,K] f16
__global__ void k_transpose_f16(const float* __restrict__ W, _Float16* __restrict__ WT,
                                int K, int N) {
  int i = blockIdx.x * 256 + threadIdx.x;
  if (i < K * N) {
    int n = i / K, k = i - n * K;
    WT[i] = (_Float16)W[(size_t)k * N + n];
  }
}

// ---------------- LDS-staged, async-pipelined TN WMMA GEMM ----------------
// C[M,N] = A[M,K](f16,row-major) * B[N,K](f16,row-major)^T  (+bias / epilogue)
// Block tile 128x128 (8 wave32s, each 32x64), k-step 32, double-buffered LDS
// filled by global_load_async_to_lds_b128 (4 x b128 issues per thread per stage).
// Per stage per wave: 6 LDS fragment loads -> 8 WMMAs.
enum { EP_F32RES = 0, EP_Q = 1, EP_K = 2, EP_VT = 3, EP_GELU = 4 };

template <int MODE>
__global__ void k_gemm_tn(const _Float16* __restrict__ A, const _Float16* __restrict__ Bm,
                          const float* __restrict__ bias, const float* __restrict__ res,
                          float* __restrict__ out32, _Float16* __restrict__ out16,
                          int M, int N, int K) {
  __shared__ _Float16 sA[2][128 * LDSP];
  __shared__ _Float16 sB[2][128 * LDSP];

  const int t    = threadIdx.x;
  const int lane = t & 31;
  const int wid  = t >> 5;                 // 8 waves: 4 (M) x 2 (N)
  const int bm0  = blockIdx.x * 128;
  const int bn0  = blockIdx.y * 128;
  const int wm0  = (wid & 3) * 32;
  const int wn0  = (wid >> 2) * 64;

  // LDS byte offsets (low 32 bits of the generic pointer == LDS offset)
  const unsigned offA[2] = {(unsigned)(uintptr_t)&sA[0][0], (unsigned)(uintptr_t)&sA[1][0]};
  const unsigned offB[2] = {(unsigned)(uintptr_t)&sB[0][0], (unsigned)(uintptr_t)&sB[1][0]};

  const int crow = t >> 2;                 // 0..63
  const int cchk = (t & 3) * 8;            // half offset of 16B chunk within 64B row

  auto issue = [&](int s, int k) {
#pragma unroll
    for (int rr = 0; rr < 2; ++rr) {       // A and B tiles: 128 rows x 64B each
      int row = rr * 64 + crow;
      unsigned loff = (unsigned)((row * LDSP + cchk) * 2);
      async_cp16(offA[s] + loff, A + (size_t)(bm0 + row) * K + k + cchk);
      async_cp16(offB[s] + loff, Bm + (size_t)(bn0 + row) * K + k + cchk);
    }
  };

  v8f acc[2][4] = {};
  issue(0, 0);
  int s = 0;
  for (int k = 0; k < K; k += 32, s ^= 1) {
    wait_async0();          // own copies done...
    __syncthreads();        // ...then everyone's copies done; also closes WAR window
    if (k + 32 < K) issue(s ^ 1, k + 32);

    v16h fa[2], fb[4];
#pragma unroll
    for (int i = 0; i < 2; ++i) fa[i] = ld_frag(sA[s], LDSP, wm0 + 16 * i, 0, lane);
#pragma unroll
    for (int j = 0; j < 4; ++j) fb[j] = ld_frag(sB[s], LDSP, wn0 + 16 * j, 0, lane);
#pragma unroll
    for (int i = 0; i < 2; ++i)
#pragma unroll
      for (int j = 0; j < 4; ++j) acc[i][j] = wmma16(fa[i], fb[j], acc[i][j]);
  }

  const int lh = (lane >> 4) * 8;   // D layout: lanes 16-31 -> M+8
  const int lc = lane & 15;

#pragma unroll
  for (int i = 0; i < 2; ++i) {
#pragma unroll
    for (int j = 0; j < 4; ++j) {
      const v8f c = acc[i][j];
#pragma unroll
      for (int r = 0; r < 8; ++r) {
        int m = bm0 + wm0 + i * 16 + lh + r;
        int n = bn0 + wn0 + j * 16 + lc;
        float v = c[r];
        if (bias) v += bias[n];
        if (MODE == EP_F32RES) {
          out32[(size_t)m * N + n] = v + res[(size_t)m * N + n];
        } else if (MODE == EP_Q) {            // [b,h,l,64], softmax scale folded in
          int b = m >> 9, l = m & 511, h = n >> 6, d = n & 63;
          out16[((((size_t)b * H_ + h) * L_ + l) << 6) + d] = (_Float16)(v * SCALE_);
        } else if (MODE == EP_K) {            // [b,h,l,32]
          int b = m >> 9, l = m & 511, h = n >> 5, d = n & 31;
          out16[((((size_t)b * H_ + h) * L_ + l) << 5) + d] = (_Float16)v;
        } else if (MODE == EP_VT) {           // transposed: [b,h,32,L]
          int b = m >> 9, l = m & 511, h = n >> 5, d = n & 31;
          out16[((((size_t)b * H_ + h) * HKV_ + d) << 9) + l] = (_Float16)v;
        } else {                              // EP_GELU, f16 row-major
          float g = 0.5f * v * (1.0f + erff(v * 0.70710678118654752f));
          out16[(size_t)m * N + n] = (_Float16)g;
        }
      }
    }
  }
}

// ---------------- attention scores: S = Q * [Ka|Kb]^T ----------------
// grid (M/128, Lkeys/64, B*H); WMMA K-step 32 == HKV so step0 reads Ka, step1 Kb.
__global__ void k_attn_scores(const _Float16* __restrict__ Q, const _Float16* __restrict__ Ka,
                              const _Float16* __restrict__ Kb, float* __restrict__ Sc) {
  const int bh = blockIdx.z;
  const _Float16* Qp  = Q  + (size_t)bh * L_ * HD_;
  const _Float16* Kap = Ka + (size_t)bh * L_ * HKV_;
  const _Float16* Kbp = Kb + (size_t)bh * L_ * HKV_;
  float* Sp = Sc + (size_t)bh * L_ * L_;

  const int lane = threadIdx.x & 31;
  const int wid  = threadIdx.x >> 5;
  const int m0 = blockIdx.x * 128 + (wid & 3) * 32;   // query
  const int n0 = blockIdx.y * 64 + (wid >> 2) * 32;   // key

  v8f a00 = {}, a01 = {}, a10 = {}, a11 = {};
  {   // d = 0..31 from modality a
    v16h fa0 = ld_frag(Qp, HD_, m0, 0, lane);
    v16h fa1 = ld_frag(Qp, HD_, m0 + 16, 0, lane);
    v16h fb0 = ld_frag(Kap, HKV_, n0, 0, lane);
    v16h fb1 = ld_frag(Kap, HKV_, n0 + 16, 0, lane);
    a00 = wmma16(fa0, fb0, a00); a01 = wmma16(fa0, fb1, a01);
    a10 = wmma16(fa1, fb0, a10); a11 = wmma16(fa1, fb1, a11);
  }
  {   // d = 32..63 from modality b
    v16h fa0 = ld_frag(Qp, HD_, m0, 32, lane);
    v16h fa1 = ld_frag(Qp, HD_, m0 + 16, 32, lane);
    v16h fb0 = ld_frag(Kbp, HKV_, n0, 0, lane);
    v16h fb1 = ld_frag(Kbp, HKV_, n0 + 16, 0, lane);
    a00 = wmma16(fa0, fb0, a00); a01 = wmma16(fa0, fb1, a01);
    a10 = wmma16(fa1, fb0, a10); a11 = wmma16(fa1, fb1, a11);
  }
  const int lh = (lane >> 4) * 8, lc = lane & 15;
  auto emit = [&](int ii, int jj, const v8f& c) {
#pragma unroll
    for (int r = 0; r < 8; ++r)
      Sp[(size_t)(m0 + ii * 16 + lh + r) * L_ + (n0 + jj * 16 + lc)] = c[r];
  };
  emit(0, 0, a00); emit(0, 1, a01); emit(1, 0, a10); emit(1, 1, a11);
}

// ---------------- softmax over keys (with optional additive mask bias) ----------------
__global__ void k_softmax(const float* __restrict__ Sc, const float* __restrict__ ma,
                          const float* __restrict__ mb, _Float16* __restrict__ P) {
  const int row = blockIdx.x;           // bh*L + q
  const int b = row >> 13;              // row / (H*L); H*L = 8192
  const float* sp = Sc + (size_t)row * L_;
  _Float16* pp = P + (size_t)row * L_;
  const int t = threadIdx.x;            // 128 threads, 4 keys each
  float vals[4], mx = -3.4e38f;
#pragma unroll
  for (int i = 0; i < 4; ++i) {
    int j = t + i * 128;
    float v = sp[j];
    if (ma) v += ma[b * L_ + j] + mb[b * L_ + j];
    vals[i] = v; mx = fmaxf(mx, v);
  }
  __shared__ float red[128];
  red[t] = mx; __syncthreads();
  for (int s = 64; s > 0; s >>= 1) { if (t < s) red[t] = fmaxf(red[t], red[t + s]); __syncthreads(); }
  mx = red[0]; __syncthreads();
  float sum = 0.f;
#pragma unroll
  for (int i = 0; i < 4; ++i) { vals[i] = expf(vals[i] - mx); sum += vals[i]; }
  red[t] = sum; __syncthreads();
  for (int s = 64; s > 0; s >>= 1) { if (t < s) red[t] += red[t + s]; __syncthreads(); }
  float inv = 1.0f / red[0];
#pragma unroll
  for (int i = 0; i < 4; ++i) pp[t + i * 128] = (_Float16)(vals[i] * inv);
}

// ---------------- ctx = P * [Va|Vb] (V stored transposed [b,h,32,L]) ----------------
__global__ void k_attn_ctx(const _Float16* __restrict__ P, const _Float16* __restrict__ Vta,
                           const _Float16* __restrict__ Vtb, _Float16* __restrict__ CTX) {
  const int bh = blockIdx.z, b = bh >> 4, h = bh & 15;
  const _Float16* Pp = P + (size_t)bh * L_ * L_;
  const _Float16* Va = Vta + (size_t)bh * HKV_ * L_;
  const _Float16* Vb = Vtb + (size_t)bh * HKV_ * L_;

  const int lane = threadIdx.x & 31;
  const int wid  = threadIdx.x >> 5;
  const int m0 = blockIdx.x * 128 + (wid & 3) * 32;
  const int n0 = (wid >> 2) * 32;                       // N = 64, grid.y == 1
  const _Float16* s0 = (n0 < 32)      ? Va + (size_t)n0 * L_        : Vb + (size_t)(n0 - 32) * L_;
  const _Float16* s1 = (n0 + 16 < 32) ? Va + (size_t)(n0 + 16) * L_ : Vb + (size_t)(n0 - 16) * L_;

  v8f a00 = {}, a01 = {}, a10 = {}, a11 = {};
  for (int k = 0; k < L_; k += 32) {
    v16h fa0 = ld_frag(Pp, L_, m0, k, lane);
    v16h fa1 = ld_frag(Pp, L_, m0 + 16, k, lane);
    v16h fb0 = ld_frag(s0, L_, 0, k, lane);
    v16h fb1 = ld_frag(s1, L_, 0, k, lane);
    a00 = wmma16(fa0, fb0, a00); a01 = wmma16(fa0, fb1, a01);
    a10 = wmma16(fa1, fb0, a10); a11 = wmma16(fa1, fb1, a11);
  }
  const int lh = (lane >> 4) * 8, lc = lane & 15;
  auto emit = [&](int ii, int jj, const v8f& c) {
#pragma unroll
    for (int r = 0; r < 8; ++r) {
      int q = m0 + ii * 16 + lh + r;
      int d = n0 + jj * 16 + lc;
      CTX[((size_t)(b * L_ + q) * BI_) + h * HD_ + d] = (_Float16)c[r];
    }
  };
  emit(0, 0, a00); emit(0, 1, a01); emit(1, 0, a10); emit(1, 1, a11);
}

// ---------------- LayerNorm over last dim (768), optional f16 mirror ----------------
__global__ void k_layernorm(const float* __restrict__ X, const float* __restrict__ g,
                            const float* __restrict__ be, float* __restrict__ O32,
                            _Float16* __restrict__ O16) {
  const int row = blockIdx.x;
  const float* xp = X + (size_t)row * DH_;
  const int t = threadIdx.x;            // 256 threads, 3 elems each
  float x[3], s = 0.f, s2 = 0.f;
#pragma unroll
  for (int i = 0; i < 3; ++i) { x[i] = xp[t + i * 256]; s += x[i]; s2 += x[i] * x[i]; }
  __shared__ float r1[256], r2[256];
  r1[t] = s; r2[t] = s2; __syncthreads();
  for (int st = 128; st > 0; st >>= 1) {
    if (t < st) { r1[t] += r1[t + st]; r2[t] += r2[t + st]; }
    __syncthreads();
  }
  float mean = r1[0] * (1.0f / DH_);
  float var  = r2[0] * (1.0f / DH_) - mean * mean;
  float inv  = rsqrtf(var + EPS_);
#pragma unroll
  for (int i = 0; i < 3; ++i) {
    int n = t + i * 256;
    float y = g[n] * (x[i] - mean) * inv + be[n];
    O32[(size_t)row * DH_ + n] = y;
    if (O16) O16[(size_t)row * DH_ + n] = (_Float16)y;
  }
}

// ---------------- host ----------------
extern "C" void kernel_launch(void* const* d_in, const int* in_sizes, int n_in,
                              void* d_out, int out_size, void* d_ws, size_t ws_size,
                              hipStream_t stream) {
  const float* x[3]    = {(const float*)d_in[0], (const float*)d_in[1], (const float*)d_in[2]};
  const float* mask[3] = {(const float*)d_in[3], (const float*)d_in[4], (const float*)d_in[5]};

  // Resolve param ordering: jax-sorted dict flatten puts Wf1 (3072*768) at idx 6;
  // insertion order puts Wq1 (768*1024) there.
  int iWq[3], iWk[3], iWv[3], iWo[3], iWi[3], iWf[3];
  int ibq[3], ibk[3], ibv[3], ibo[3], ibi[3], ibf[3];
  int ig[3], ibe[3], ig2[3], ibe2[3];
  bool sorted = (n_in > 6) && (in_sizes[6] == IM_ * DH_);
  if (sorted) {
    for (int m = 0; m < 3; ++m) {
      iWf[m] = 6 + m;  iWi[m] = 9 + m;  iWk[m] = 12 + m;
      iWo[m] = 15 + m; iWq[m] = 18 + m; iWv[m] = 21 + m;
      ibf[m] = 30 + m; ibi[m] = 33 + m; ibk[m] = 36 + m;
      ibo[m] = 39 + m; ibq[m] = 42 + m; ibv[m] = 45 + m;
    }
    ibe[0] = 24; ibe[1] = 25; ibe[2] = 29;  ibe2[0] = 26; ibe2[1] = 27; ibe2[2] = 28;
    ig[0] = 48;  ig[1] = 49;  ig[2] = 53;   ig2[0] = 50;  ig2[1] = 51;  ig2[2] = 52;
  } else {
    for (int m = 0; m < 3; ++m) {
      int b0 = 6 + 16 * m;
      iWq[m] = b0;      ibq[m] = b0 + 1;  iWk[m] = b0 + 2;  ibk[m] = b0 + 3;
      iWv[m] = b0 + 4;  ibv[m] = b0 + 5;  iWo[m] = b0 + 6;  ibo[m] = b0 + 7;
      ig[m]  = b0 + 8;  ibe[m] = b0 + 9;  iWi[m] = b0 + 10; ibi[m] = b0 + 11;
      iWf[m] = b0 + 12; ibf[m] = b0 + 13; ig2[m] = b0 + 14; ibe2[m] = b0 + 15;
    }
  }
  auto PP = [&](int i) { return (const float*)d_in[i]; };

  // workspace carve-out
  size_t off = 0;
  auto alloc = [&](size_t bytes) -> void* {
    off = (off + 255) & ~(size_t)255;
    void* p = (char*)d_ws + off;
    off += bytes;
    return p;
  };
  _Float16 *Xh[3], *Qh[3], *Kh[3], *Vth[3];
  _Float16 *WqT[3], *WkT[3], *WvT[3], *WoT[3], *WiT[3], *WfT[3];
  for (int m = 0; m < 3; ++m) {
    Xh[m]  = (_Float16*)alloc((size_t)NTOK * DH_ * 2);
    Qh[m]  = (_Float16*)alloc((size_t)NTOK * BI_ * 2);
    Kh[m]  = (_Float16*)alloc((size_t)NTOK * KV_ * 2);
    Vth[m] = (_Float16*)alloc((size_t)NTOK * KV_ * 2);
    WqT[m] = (_Float16*)alloc((size_t)BI_ * DH_ * 2);
    WkT[m] = (_Float16*)alloc((size_t)KV_ * DH_ * 2);
    WvT[m] = (_Float16*)alloc((size_t)KV_ * DH_ * 2);
    WoT[m] = (_Float16*)alloc((size_t)DH_ * BI_ * 2);
    WiT[m] = (_Float16*)alloc((size_t)IM_ * DH_ * 2);
    WfT[m] = (_Float16*)alloc((size_t)DH_ * IM_ * 2);
  }
  float*     SC    = (float*)alloc((size_t)B_ * H_ * L_ * L_ * 4);
  _Float16*  Pm    = (_Float16*)alloc((size_t)B_ * H_ * L_ * L_ * 2);
  _Float16*  CTX   = (_Float16*)alloc((size_t)NTOK * BI_ * 2);
  float*     PRE   = (float*)alloc((size_t)NTOK * DH_ * 4);
  float*     ATT32 = (float*)alloc((size_t)NTOK * DH_ * 4);
  _Float16*  ATT16 = (_Float16*)alloc((size_t)NTOK * DH_ * 2);
  _Float16*  INTER = (_Float16*)alloc((size_t)NTOK * IM_ * 2);

  dim3 blk(256);

  // activations + weights -> f16 (weights transposed to [N,K])
  for (int m = 0; m < 3; ++m) {
    k_f32_to_f16<<<(NTOK * DH_ + 255) / 256, blk, 0, stream>>>(x[m], Xh[m], NTOK * DH_);
    k_transpose_f16<<<(DH_ * BI_ + 255) / 256, blk, 0, stream>>>(PP(iWq[m]), WqT[m], DH_, BI_);
    k_transpose_f16<<<(DH_ * KV_ + 255) / 256, blk, 0, stream>>>(PP(iWk[m]), WkT[m], DH_, KV_);
    k_transpose_f16<<<(DH_ * KV_ + 255) / 256, blk, 0, stream>>>(PP(iWv[m]), WvT[m], DH_, KV_);
    k_transpose_f16<<<(BI_ * DH_ + 255) / 256, blk, 0, stream>>>(PP(iWo[m]), WoT[m], BI_, DH_);
    k_transpose_f16<<<(DH_ * IM_ + 255) / 256, blk, 0, stream>>>(PP(iWi[m]), WiT[m], DH_, IM_);
    k_transpose_f16<<<(IM_ * DH_ + 255) / 256, blk, 0, stream>>>(PP(iWf[m]), WfT[m], IM_, DH_);
  }

  // Q/K/V projections (Q carries softmax scale; V stored transposed per head)
  // grids: (M/128, N/128)
  for (int m = 0; m < 3; ++m) {
    k_gemm_tn<EP_Q ><<<dim3(16, 8), blk, 0, stream>>>(Xh[m], WqT[m], PP(ibq[m]), nullptr,
                                                      nullptr, Qh[m], NTOK, BI_, DH_);
    k_gemm_tn<EP_K ><<<dim3(16, 4), blk, 0, stream>>>(Xh[m], WkT[m], PP(ibk[m]), nullptr,
                                                      nullptr, Kh[m], NTOK, KV_, DH_);
    k_gemm_tn<EP_VT><<<dim3(16, 4), blk, 0, stream>>>(Xh[m], WvT[m], PP(ibv[m]), nullptr,
                                                      nullptr, Vth[m], NTOK, KV_, DH_);
  }

  // branches: (query modality, kv pair, mask?)
  const int ka[3] = {1, 0, 0}, kb[3] = {2, 2, 1};
  const bool hasmask[3] = {false, true, true};
  for (int br = 0; br < 3; ++br) {
    const int qi = br, a = ka[br], bb = kb[br];
    k_attn_scores<<<dim3(4, 8, B_ * H_), blk, 0, stream>>>(Qh[qi], Kh[a], Kh[bb], SC);
    k_softmax<<<dim3(B_ * H_ * L_), dim3(128), 0, stream>>>(
        SC, hasmask[br] ? mask[a] : nullptr, hasmask[br] ? mask[bb] : nullptr, Pm);
    k_attn_ctx<<<dim3(4, 1, B_ * H_), blk, 0, stream>>>(Pm, Vth[a], Vth[bb], CTX);
    // att = LN(ctx@Wo + bo + x)
    k_gemm_tn<EP_F32RES><<<dim3(16, 6), blk, 0, stream>>>(CTX, WoT[qi], PP(ibo[qi]), x[qi],
                                                          PRE, nullptr, NTOK, DH_, BI_);
    k_layernorm<<<dim3(NTOK), dim3(256), 0, stream>>>(PRE, PP(ig[qi]), PP(ibe[qi]), ATT32, ATT16);
    // inter = gelu(att@Wi + bi)
    k_gemm_tn<EP_GELU><<<dim3(16, 24), blk, 0, stream>>>(ATT16, WiT[qi], PP(ibi[qi]), nullptr,
                                                         nullptr, INTER, NTOK, IM_, DH_);
    // out = LN(inter@Wf + bf + att)
    k_gemm_tn<EP_F32RES><<<dim3(16, 6), blk, 0, stream>>>(INTER, WfT[qi], PP(ibf[qi]), ATT32,
                                                          PRE, nullptr, NTOK, DH_, IM_);
    k_layernorm<<<dim3(NTOK), dim3(256), 0, stream>>>(
        PRE, PP(ig2[qi]), PP(ibe2[qi]),
        (float*)d_out + (size_t)qi * NTOK * DH_, nullptr);
  }
}
